// Block_85633057947963
// MI455X (gfx1250) — compile-verified, matrix operand order checked by reading, not run on hardware
//
#include <hip/hip_runtime.h>

// ---------------------------------------------------------------------------
// Types for CDNA5 WMMA (wave32)
// ---------------------------------------------------------------------------
typedef __attribute__((ext_vector_type(16))) __bf16 v16bf;
typedef __attribute__((ext_vector_type(8)))  __bf16 v8bf;
typedef __attribute__((ext_vector_type(8)))  float  v8f;

#define BN_EPS 1e-5f

// Problem dims (fixed by the reference)
#define BB   16
#define NN   1024
#define CC   512
#define HH   8
#define HD   64
#define HID  2048
#define MTOT (BB * NN)   // 16384 rows

// ---------------------------------------------------------------------------
// Fragment helpers
//
// 16-bit A/B operand layout (ISA 7.12.2): lanes 0-15 hold row (lane&15) with
// K = {0..7, 16..23}; lanes 16-31 hold the same rows with K = {8..15, 24..31}.
// So per lane: two contiguous 16-byte loads at +0 and +16 elements, with an
// extra +8 element offset for the upper half-wave.
// ---------------------------------------------------------------------------
static __device__ __forceinline__ v16bf load_frag(const __bf16* p) {
  v8bf lo = *(const v8bf*)(p);
  v8bf hi = *(const v8bf*)(p + 16);
  return __builtin_shufflevector(lo, hi, 0, 1, 2, 3, 4, 5, 6, 7,
                                         8, 9, 10, 11, 12, 13, 14, 15);
}

static __device__ __forceinline__ v8f wmma_bf16(v16bf a, v16bf b, v8f c) {
  // D = A(16x32) * B(32x16) + C, f32 accumulate
  return __builtin_amdgcn_wmma_f32_16x16x32_bf16(
      /*neg_a=*/false, a, /*neg_b=*/false, b,
      /*c_mod=*/(short)0, c, /*reuse_a=*/false, /*reuse_b=*/false);
}

// ---------------------------------------------------------------------------
// BN folding:  y = relu( x @ (s*W)^T + (s*bias + be - s*m) ),  s = g*rsqrt(v+eps)
// ---------------------------------------------------------------------------
__global__ void fold_bn(const float* __restrict__ W, const float* __restrict__ bias,
                        const float* __restrict__ g, const float* __restrict__ be,
                        const float* __restrict__ m, const float* __restrict__ v,
                        __bf16* __restrict__ Wb, float* __restrict__ fb,
                        int outF, int inF) {
  int idx = blockIdx.x * 256 + threadIdx.x;
  int total = outF * inF;
  if (idx < total) {
    int o = idx / inF;
    float s = g[o] * rsqrtf(v[o] + BN_EPS);
    Wb[idx] = (__bf16)(W[idx] * s);
  }
  if (idx < outF) {
    float s = g[idx] * rsqrtf(v[idx] + BN_EPS);
    fb[idx] = bias[idx] * s + be[idx] - m[idx] * s;
  }
}

__global__ void f32_to_bf16(const float* __restrict__ in, __bf16* __restrict__ out,
                            int n) {
  int idx = blockIdx.x * 256 + threadIdx.x;
  if (idx < n) out[idx] = (__bf16)in[idx];
}

// ---------------------------------------------------------------------------
// Generic fused GEMM:  out = relu(A[MxK] @ W[NxK]^T + fbias) (+ resid)
// Block: 256 threads = 8 waves arranged 4(M) x 2(N); wave tile 32x64;
// block tile 128x128; K stepped by 32 with v_wmma_f32_16x16x32_bf16.
// ---------------------------------------------------------------------------
__global__ void __launch_bounds__(256)
gemm_bn_relu(const __bf16* __restrict__ A, const __bf16* __restrict__ W,
             const float* __restrict__ fbias, const float* __restrict__ resid,
             float* __restrict__ outF, __bf16* __restrict__ outB,
             int M, int N, int K) {
  const int lane  = threadIdx.x & 31;
  const int wid   = threadIdx.x >> 5;
  const int waveM = wid & 3;
  const int waveN = wid >> 2;
  const int mTile = blockIdx.y * 128 + waveM * 32;
  const int nTile = blockIdx.x * 128 + waveN * 64;
  const int rowA  = lane & 15;
  const int koff  = (lane >> 4) * 8;

  v8f acc[2][4];
#pragma unroll
  for (int mi = 0; mi < 2; mi++)
#pragma unroll
    for (int ni = 0; ni < 4; ni++)
#pragma unroll
      for (int r = 0; r < 8; r++) acc[mi][ni][r] = 0.0f;

  const __bf16* aRow0 = A + (size_t)(mTile + rowA) * K + koff;
  const __bf16* aRow1 = aRow0 + (size_t)16 * K;
  const __bf16* wRow[4];
#pragma unroll
  for (int ni = 0; ni < 4; ni++)
    wRow[ni] = W + (size_t)(nTile + ni * 16 + rowA) * K + koff;

  for (int k0 = 0; k0 < K; k0 += 32) {
    // speculative prefetch of the next K tile (global_prefetch_b8)
    __builtin_prefetch(aRow0 + k0 + 32, 0, 1);
    __builtin_prefetch(wRow[0] + k0 + 32, 0, 1);

    v16bf a0 = load_frag(aRow0 + k0);
    v16bf a1 = load_frag(aRow1 + k0);
#pragma unroll
    for (int ni = 0; ni < 4; ni++) {
      v16bf b = load_frag(wRow[ni] + k0);
      acc[0][ni] = wmma_bf16(a0, b, acc[0][ni]);
      acc[1][ni] = wmma_bf16(a1, b, acc[1][ni]);
    }
  }

  // Epilogue: C/D layout = 8 VGPRs; lane l, reg r -> M = r + 8*(l>=16), N = l&15
  const int mBase = mTile + (lane >> 4) * 8;
  const int nLane = lane & 15;
#pragma unroll
  for (int mi = 0; mi < 2; mi++) {
#pragma unroll
    for (int ni = 0; ni < 4; ni++) {
      const int nCol = nTile + ni * 16 + nLane;
      const float fb = fbias ? fbias[nCol] : 0.0f;
#pragma unroll
      for (int r = 0; r < 8; r++) {
        const int mRow = mBase + mi * 16 + r;
        float val = acc[mi][ni][r] + fb;
        val = fmaxf(val, 0.0f);                       // fused ReLU
        const size_t idx = (size_t)mRow * N + nCol;
        if (resid) val += resid[idx];                 // residual after ReLU
        if (outF) outF[idx] = val;
        if (outB) outB[idx] = (__bf16)val;
      }
    }
  }
}

// ---------------------------------------------------------------------------
// KV^T kernel: per (b,h), KVt[d2][d1] = 0.125 * sum_n v[n,d2] * k[n,d1]
// (linear attention associativity). One workgroup per (b,h); LDS-staged.
// ---------------------------------------------------------------------------
__global__ void __launch_bounds__(256)
kv_kernel(const __bf16* __restrict__ k, const __bf16* __restrict__ v,
          __bf16* __restrict__ kvt) {
  const int bh = blockIdx.x;
  const int b  = bh >> 3;
  const int h  = bh & 7;
  const __bf16* kp = k + (size_t)b * NN * CC + h * HD;  // rows n, stride CC
  const __bf16* vp = v + (size_t)b * NN * CC + h * HD;

  __shared__ __bf16 sk[64][64];
  __shared__ __bf16 sv[64][64];

  const int d2 = threadIdx.x & 63;        // output column (v side)
  const int g  = threadIdx.x >> 6;        // d1 group: rows g*16..g*16+15
  float acc[16];
#pragma unroll
  for (int i = 0; i < 16; i++) acc[i] = 0.0f;

  const int lr = threadIdx.x >> 2;        // cooperative load row 0..63
  const int lc = (threadIdx.x & 3) * 16;  // cooperative load col segment

  for (int n0 = 0; n0 < NN; n0 += 64) {
    const __bf16* krow = kp + (size_t)(n0 + lr) * CC + lc;
    const __bf16* vrow = vp + (size_t)(n0 + lr) * CC + lc;
    *(v8bf*)&sk[lr][lc]     = *(const v8bf*)(krow);
    *(v8bf*)&sk[lr][lc + 8] = *(const v8bf*)(krow + 8);
    *(v8bf*)&sv[lr][lc]     = *(const v8bf*)(vrow);
    *(v8bf*)&sv[lr][lc + 8] = *(const v8bf*)(vrow + 8);
    __syncthreads();
#pragma unroll 4
    for (int n = 0; n < 64; n++) {
      const float vv = (float)sv[n][d2];
      v8bf k0v = *(const v8bf*)&sk[n][g * 16];
      v8bf k1v = *(const v8bf*)&sk[n][g * 16 + 8];
#pragma unroll
      for (int i = 0; i < 8; i++) {
        acc[i]     += vv * (float)k0v[i];
        acc[i + 8] += vv * (float)k1v[i];
      }
    }
    __syncthreads();
  }

  __bf16* out = kvt + (size_t)bh * HD * HD + (size_t)d2 * HD + g * 16;
#pragma unroll
  for (int i = 0; i < 16; i++) out[i] = (__bf16)(acc[i] * 0.125f);
}

// ---------------------------------------------------------------------------
// Attention apply: o[b,:,h*64:+64] = relu( q[b,:,h*64:+64] @ KVt[bh]^T )
// M=1024 per (b,h), K=N=64. 8 waves stacked in M -> block tile 256x64.
// ---------------------------------------------------------------------------
__global__ void __launch_bounds__(256)
attn_gemm(const __bf16* __restrict__ q, const __bf16* __restrict__ kvt,
          __bf16* __restrict__ o) {
  const int bh = blockIdx.y;
  const int b  = bh >> 3;
  const int h  = bh & 7;
  const __bf16* A  = q + (size_t)b * NN * CC + h * HD;     // lda = CC
  const __bf16* Wk = kvt + (size_t)bh * HD * HD;           // 64x64, ld = 64
  __bf16* O        = o + (size_t)b * NN * CC + h * HD;     // ldc = CC

  const int lane  = threadIdx.x & 31;
  const int wid   = threadIdx.x >> 5;
  const int mTile = blockIdx.x * 256 + wid * 32;
  const int rowA  = lane & 15;
  const int koff  = (lane >> 4) * 8;

  v8f acc[2][4];
#pragma unroll
  for (int mi = 0; mi < 2; mi++)
#pragma unroll
    for (int ni = 0; ni < 4; ni++)
#pragma unroll
      for (int r = 0; r < 8; r++) acc[mi][ni][r] = 0.0f;

  const __bf16* aRow0 = A + (size_t)(mTile + rowA) * CC + koff;
  const __bf16* aRow1 = aRow0 + (size_t)16 * CC;

#pragma unroll
  for (int k0 = 0; k0 < HD; k0 += 32) {
    v16bf a0 = load_frag(aRow0 + k0);
    v16bf a1 = load_frag(aRow1 + k0);
#pragma unroll
    for (int ni = 0; ni < 4; ni++) {
      v16bf bfr = load_frag(Wk + (size_t)(ni * 16 + rowA) * HD + koff + k0);
      acc[0][ni] = wmma_bf16(a0, bfr, acc[0][ni]);
      acc[1][ni] = wmma_bf16(a1, bfr, acc[1][ni]);
    }
  }

  const int mBase = mTile + (lane >> 4) * 8;
  const int nLane = lane & 15;
#pragma unroll
  for (int mi = 0; mi < 2; mi++)
#pragma unroll
    for (int ni = 0; ni < 4; ni++)
#pragma unroll
      for (int r = 0; r < 8; r++) {
        const int mRow = mBase + mi * 16 + r;
        const int nCol = ni * 16 + nLane;
        float val = fmaxf(acc[mi][ni][r], 0.0f);      // attn_lif ReLU
        O[(size_t)mRow * CC + nCol] = (__bf16)val;
      }
}

// ---------------------------------------------------------------------------
// Host-side orchestration
// ---------------------------------------------------------------------------
extern "C" void kernel_launch(void* const* d_in, const int* in_sizes, int n_in,
                              void* d_out, int out_size, void* d_ws, size_t ws_size,
                              hipStream_t stream) {
  (void)in_sizes; (void)n_in; (void)out_size; (void)ws_size;

  const float* x = (const float*)d_in[0];
  // input layout: x, then 6 params per stage for q,k,v,p,f1,f2
  const float* P[6][6];
  for (int s = 0; s < 6; s++)
    for (int j = 0; j < 6; j++)
      P[s][j] = (const float*)d_in[1 + s * 6 + j];
  const int outFs[6] = {CC, CC, CC, CC, HID, CC};
  const int inFs[6]  = {CC, CC, CC, CC, CC, HID};

  // ---- workspace bump allocator ----
  char* ws = (char*)d_ws;
  size_t off = 0;
  auto alloc = [&](size_t bytes) -> void* {
    off = (off + 255) & ~(size_t)255;
    void* p = ws + off;
    off += bytes;
    return p;
  };

  __bf16* xb  = (__bf16*)alloc((size_t)MTOT * CC * 2);
  __bf16* qb  = (__bf16*)alloc((size_t)MTOT * CC * 2);
  __bf16* kb  = (__bf16*)alloc((size_t)MTOT * CC * 2);
  __bf16* vb  = (__bf16*)alloc((size_t)MTOT * CC * 2);
  __bf16* ob  = (__bf16*)alloc((size_t)MTOT * CC * 2);
  __bf16* x1b = (__bf16*)alloc((size_t)MTOT * CC * 2);
  float*  x1f = (float*) alloc((size_t)MTOT * CC * 4);
  __bf16* hb  = (__bf16*)alloc((size_t)MTOT * HID * 2);
  __bf16* kvt = (__bf16*)alloc((size_t)BB * HH * HD * HD * 2);
  __bf16* Wb[6];
  float*  fb[6];
  for (int s = 0; s < 6; s++) {
    Wb[s] = (__bf16*)alloc((size_t)outFs[s] * inFs[s] * 2);
    fb[s] = (float*) alloc((size_t)outFs[s] * 4);
  }

  // ---- 1. fold BN into weights (bf16) + fused bias ----
  for (int s = 0; s < 6; s++) {
    int total = outFs[s] * inFs[s];
    fold_bn<<<(total + 255) / 256, 256, 0, stream>>>(
        P[s][0], P[s][1], P[s][2], P[s][3], P[s][4], P[s][5],
        Wb[s], fb[s], outFs[s], inFs[s]);
  }

  // ---- 2. x -> bf16 ----
  {
    int n = MTOT * CC;
    f32_to_bf16<<<(n + 255) / 256, 256, 0, stream>>>(x, xb, n);
  }

  // ---- 3. q/k/v stages: relu(BN(x@W^T+b)) -> bf16 ----
  dim3 blk(256);
  dim3 grdC(CC / 128, MTOT / 128);     // N=512
  gemm_bn_relu<<<grdC, blk, 0, stream>>>(xb, Wb[0], fb[0], nullptr,
                                         nullptr, qb, MTOT, CC, CC);
  gemm_bn_relu<<<grdC, blk, 0, stream>>>(xb, Wb[1], fb[1], nullptr,
                                         nullptr, kb, MTOT, CC, CC);
  gemm_bn_relu<<<grdC, blk, 0, stream>>>(xb, Wb[2], fb[2], nullptr,
                                         nullptr, vb, MTOT, CC, CC);

  // ---- 4. KVt = 0.125 * k^T v per (b,h)  (linear-attention associativity) ----
  kv_kernel<<<BB * HH, 256, 0, stream>>>(kb, vb, kvt);

  // ---- 5. o = relu(q @ KVt^T) per head -> bf16 ----
  attn_gemm<<<dim3(NN / 256, BB * HH), blk, 0, stream>>>(qb, kvt, ob);

  // ---- 6. x1 = x + stage_p(o): emit f32 (for residual) and bf16 (for f1) ----
  gemm_bn_relu<<<grdC, blk, 0, stream>>>(ob, Wb[3], fb[3], x,
                                         x1f, x1b, MTOT, CC, CC);

  // ---- 7. h = stage_f1(x1): 512 -> 2048 ----
  dim3 grdH(HID / 128, MTOT / 128);
  gemm_bn_relu<<<grdH, blk, 0, stream>>>(x1b, Wb[4], fb[4], nullptr,
                                         nullptr, hb, MTOT, HID, CC);

  // ---- 8. y = x1 + stage_f2(h): 2048 -> 512, write final f32 output ----
  gemm_bn_relu<<<grdC, blk, 0, stream>>>(hb, Wb[5], fb[5], x1f,
                                         (float*)d_out, nullptr, MTOT, CC, HID);
}
